// GraphEncoderLayer_31928786879008
// MI455X (gfx1250) — compile-verified, matrix operand order checked by reading, not run on hardware
//
#include <hip/hip_runtime.h>
#include <math.h>

#define T_DIM 1024
#define B_DIM 8
#define C_DIM 256
#define H_DIM 8
#define D_DIM 32
#define F_DIM 512
#define M_DIM (T_DIM * B_DIM)   // 8192 flattened (t*B+b) rows

typedef __bf16 bf16;
typedef __attribute__((ext_vector_type(16))) __bf16        v16bf;
typedef __attribute__((ext_vector_type(8)))  float         v8f;
typedef __attribute__((ext_vector_type(4)))  unsigned int  u32x4;
typedef __attribute__((ext_vector_type(4)))  float         f32x4;

union Frag  { v16bf v; u32x4 q[2]; };
union Pack8 { u32x4 q; bf16 h[8]; };

// A/B fragment loader for v_wmma_f32_16x16x32_bf16.
// 16-bit A (16x32) layout: lane = m + 16*hi holds K = {hi*8..+7, 16+hi*8..+7}
// (CDNA5 ISA); B mirrors with lane = n + 16*hi. Operands stored
// K-contiguous => two global_load_b128 per lane.
__device__ __forceinline__ v16bf load_tile(const bf16* __restrict__ base, int stride,
                                           int r0, int k0, int lane) {
  const int ln = lane & 15, hi = lane >> 4;
  Frag f;
  const bf16* p = base + (size_t)(r0 + ln) * stride + k0 + hi * 8;
  f.q[0] = *(const u32x4*)p;
  f.q[1] = *(const u32x4*)(p + 16);
  return f.v;
}

__device__ __forceinline__ v8f wmma_bf16(v16bf a, v16bf b, v8f c) {
  return __builtin_amdgcn_wmma_f32_16x16x32_bf16(false, a, false, b, (short)0, c,
                                                 false, false);
}

// ---------------------------------------------------------------- fp32->bf16
__global__ void __launch_bounds__(256) cvt_kernel(const float* __restrict__ in,
                                                  bf16* __restrict__ out, int n) {
  int i = blockIdx.x * 256 + threadIdx.x;
  if (i < n) out[i] = (bf16)in[i];
}

// ---------------------------------------------------------------- LayerNorm
// One wave per 256-wide row; lane owns 8 consecutive floats.
__global__ void __launch_bounds__(256)
ln_kernel(const float* __restrict__ x, const float* __restrict__ g,
          const float* __restrict__ bb, bf16* __restrict__ y) {
  int lane = threadIdx.x & 31, wave = threadIdx.x >> 5;
  int row = blockIdx.x * 8 + wave;
  const float* xr = x + (size_t)row * C_DIM + lane * 8;
  f32x4 a = *(const f32x4*)xr;
  f32x4 c = *(const f32x4*)(xr + 4);
  float vals[8] = {a[0], a[1], a[2], a[3], c[0], c[1], c[2], c[3]};
  float s = 0.f, ss = 0.f;
#pragma unroll
  for (int r = 0; r < 8; r++) { s += vals[r]; ss += vals[r] * vals[r]; }
#pragma unroll
  for (int m = 1; m < 32; m <<= 1) {
    s += __shfl_xor(s, m, 32);
    ss += __shfl_xor(ss, m, 32);
  }
  float mu = s * (1.f / 256.f);
  float var = ss * (1.f / 256.f) - mu * mu;
  float inv = rsqrtf(var + 1e-5f);
  Pack8 pk;
  const float* gp = g + lane * 8;
  const float* bp = bb + lane * 8;
#pragma unroll
  for (int r = 0; r < 8; r++)
    pk.h[r] = (bf16)((vals[r] - mu) * inv * gp[r] + bp[r]);
  *(u32x4*)(y + (size_t)row * C_DIM + lane * 8) = pk.q;
}

// ---------------------------------------------------------------- GEMM
// out[m,n] = sum_k A[m,k] * B[n,k] + bias[n]  (A: MxK bf16, B: NxK bf16)
// MODE 0: scatter into q/k/v buffers (V transposed to [BH, D, T])
// MODE 1: outF = acc + bias + resid   (fp32)
// MODE 2: outB = gelu(acc + bias)     (bf16)
template <int MODE>
__global__ void __launch_bounds__(128)
gemm_bf16(const bf16* __restrict__ A, const bf16* __restrict__ Bm,
          const float* __restrict__ bias, const float* __restrict__ resid,
          float* __restrict__ outF, bf16* __restrict__ outB,
          bf16* __restrict__ qb, bf16* __restrict__ kbuf, bf16* __restrict__ vb,
          int N, int K) {
  int lane = threadIdx.x & 31, wave = threadIdx.x >> 5;
  int mbase = blockIdx.y * 64 + (wave >> 1) * 32;
  int nbase = blockIdx.x * 64 + (wave & 1) * 32;
  v8f c[2][2] = {};
  for (int k0 = 0; k0 < K; k0 += 32) {
    v16bf a0 = load_tile(A, K, mbase, k0, lane);
    v16bf a1 = load_tile(A, K, mbase + 16, k0, lane);
    v16bf b0 = load_tile(Bm, K, nbase, k0, lane);
    v16bf b1 = load_tile(Bm, K, nbase + 16, k0, lane);
    c[0][0] = wmma_bf16(a0, b0, c[0][0]);
    c[0][1] = wmma_bf16(a0, b1, c[0][1]);
    c[1][0] = wmma_bf16(a1, b0, c[1][0]);
    c[1][1] = wmma_bf16(a1, b1, c[1][1]);
  }
  int ln = lane & 15, hi = lane >> 4;
#pragma unroll
  for (int i = 0; i < 2; i++)
#pragma unroll
    for (int j = 0; j < 2; j++) {
#pragma unroll
      for (int r = 0; r < 8; r++) {
        int row = mbase + i * 16 + r + 8 * hi;   // C-layout: VGPR r, hi half
        int col = nbase + j * 16 + ln;
        float v = c[i][j][r] + bias[col];
        if (MODE == 0) {
          int sel = col >> 8, cc = col & 255;
          int h = cc >> 5, d = cc & 31;
          int t = row >> 3, b = row & 7;           // row = t*B + b
          int bh = b * H_DIM + h;
          if (sel == 0)
            qb[((size_t)bh * T_DIM + t) * D_DIM + d] = (bf16)v;
          else if (sel == 1)
            kbuf[((size_t)bh * T_DIM + t) * D_DIM + d] = (bf16)v;
          else
            vb[((size_t)bh * D_DIM + d) * T_DIM + t] = (bf16)v;  // transposed
        } else if (MODE == 1) {
          outF[(size_t)row * N + col] = v + resid[(size_t)row * N + col];
        } else {
          float gl = 0.5f * v * (1.0f + erff(v * 0.70710678118654752f));
          outB[(size_t)row * N + col] = (bf16)gl;
        }
      }
    }
}

// ---------------------------------------------------------------- attention
// One wave per (b,h, 16-row Q tile); streaming softmax over K in steps of 64.
// Edge-bias (256 MB, read-once) is loaded non-temporally and branchlessly.
// Row-sums of P are accumulated with a WMMA against an all-ones B fragment
// (same C-layout rows as the O accumulators), removing the sum butterfly.
__global__ void __launch_bounds__(128)
attn_kernel(const bf16* __restrict__ Q, const bf16* __restrict__ Km,
            const bf16* __restrict__ V, const float* __restrict__ eb,
            const unsigned char* __restrict__ kpm, bf16* __restrict__ O) {
  __shared__ __attribute__((aligned(16))) bf16 pbuf[4][16 * 64];
  int lane = threadIdx.x & 31, wave = threadIdx.x >> 5;
  int task = blockIdx.x * 4 + wave;
  int bh = task >> 6;             // 64 q-tiles per (b,h)
  int qbase = (task & 63) * 16;
  int b = bh >> 3, h = bh & 7;
  int ln = lane & 15, hi = lane >> 4;
  const float scale = 0.17677669529663687f;  // 1/sqrt(32)

  const bf16* qbase_p = Q + (size_t)bh * T_DIM * D_DIM;
  const bf16* kbase_p = Km + (size_t)bh * T_DIM * D_DIM;
  const bf16* vbase_p = V + (size_t)bh * D_DIM * T_DIM;
  const float* ebb = eb + (size_t)bh * T_DIM * T_DIM;

  v16bf qa = load_tile(qbase_p, D_DIM, qbase, 0, lane);  // 16x32 = full D
  v16bf ones;
#pragma unroll
  for (int i = 0; i < 16; i++) ones[i] = (bf16)1.0f;     // all-ones 32x16 B tile
  float rm[8];
#pragma unroll
  for (int r = 0; r < 8; r++) rm[r] = -3.0e38f;
  v8f o0 = {}, o1 = {}, osum = {}, zero = {};

  for (int kb = 0; kb < T_DIM; kb += 64) {
    // ---- scores for 64 key columns: 4 WMMAs
    v8f s[4];
#pragma unroll
    for (int j = 0; j < 4; j++) {
      v16bf kt = load_tile(kbase_p, D_DIM, kb + j * 16, 0, lane);
      s[j] = wmma_bf16(qa, kt, zero);
    }
    // ---- V fragments (independent of scores): issue early for overlap
    v16bf vf[2][2];
#pragma unroll
    for (int ks = 0; ks < 2; ks++) {
      vf[ks][0] = load_tile(vbase_p, T_DIM, 0, kb + ks * 32, lane);
      vf[ks][1] = load_tile(vbase_p, T_DIM, 16, kb + ks * 32, lane);
    }
    // ---- padding mask (branchless select later)
    int col = kb + ln;
    bool msk[4];
#pragma unroll
    for (int j = 0; j < 4; j++) msk[j] = kpm[b * T_DIM + col + j * 16] != 0;
    // ---- bias add + online softmax per C-layout row r
#pragma unroll
    for (int r = 0; r < 8; r++) {
      int row = qbase + r + 8 * hi;
      const float* bp = ebb + (size_t)row * T_DIM + col;
      float v[4];
#pragma unroll
      for (int j = 0; j < 4; j++) {
        float bias = __builtin_nontemporal_load(bp + j * 16);  // streamed once
        v[j] = msk[j] ? -1e9f : fmaf(s[j][r], scale, bias);
      }
      float tmax = fmaxf(fmaxf(v[0], v[1]), fmaxf(v[2], v[3]));
#pragma unroll
      for (int m = 1; m < 16; m <<= 1) tmax = fmaxf(tmax, __shfl_xor(tmax, m, 32));
      float mnew = fmaxf(rm[r], tmax);
      float corr = __expf(rm[r] - mnew);
      rm[r] = mnew;
      o0[r] *= corr;
      o1[r] *= corr;
      osum[r] *= corr;
      // C-layout -> row-major 16x64 P tile in LDS (per-wave slice)
#pragma unroll
      for (int j = 0; j < 4; j++)
        pbuf[wave][(r + 8 * hi) * 64 + j * 16 + ln] = (bf16)__expf(v[j] - mnew);
    }
    asm volatile("s_wait_dscnt 0" ::: "memory");
    // ---- O += P * V, rowsum += P * ones  (two 16x32 P fragments)
#pragma unroll
    for (int ks = 0; ks < 2; ks++) {
      Frag pf;
      const bf16* pp = &pbuf[wave][ln * 64 + ks * 32 + hi * 8];
      pf.q[0] = *(const u32x4*)pp;
      pf.q[1] = *(const u32x4*)(pp + 16);
      o0 = wmma_bf16(pf.v, vf[ks][0], o0);
      o1 = wmma_bf16(pf.v, vf[ks][1], o1);
      osum = wmma_bf16(pf.v, ones, osum);
    }
  }
#pragma unroll
  for (int r = 0; r < 8; r++) {
    int t = qbase + r + 8 * hi;
    float inv = 1.0f / osum[r];   // every column of osum holds the row sum
    size_t base = ((size_t)t * B_DIM + b) * C_DIM + h * D_DIM;
    O[base + ln] = (bf16)(o0[r] * inv);
    O[base + 16 + ln] = (bf16)(o1[r] * inv);
  }
}

// ---------------------------------------------------------------- launch
extern "C" void kernel_launch(void* const* d_in, const int* in_sizes, int n_in,
                              void* d_out, int out_size, void* d_ws, size_t ws_size,
                              hipStream_t stream) {
  const float* x      = (const float*)d_in[0];
  const unsigned char* kpm = (const unsigned char*)d_in[1];
  const float* edge   = (const float*)d_in[2];
  const float* ln1_g  = (const float*)d_in[3];
  const float* ln1_b  = (const float*)d_in[4];
  const float* ln2_g  = (const float*)d_in[5];
  const float* ln2_b  = (const float*)d_in[6];
  const float* w_qkv  = (const float*)d_in[7];
  const float* b_qkv  = (const float*)d_in[8];
  const float* w_o    = (const float*)d_in[9];
  const float* b_o    = (const float*)d_in[10];
  const float* w1     = (const float*)d_in[11];
  const float* b1     = (const float*)d_in[12];
  const float* w2     = (const float*)d_in[13];
  const float* b2     = (const float*)d_in[14];
  float* out = (float*)d_out;

  char* p = (char*)d_ws;
  auto take = [&](size_t bytes) { char* r = p; p += (bytes + 255) & ~(size_t)255; return r; };
  bf16* wqkv_b = (bf16*)take((size_t)3 * C_DIM * C_DIM * 2);
  bf16* wo_b   = (bf16*)take((size_t)C_DIM * C_DIM * 2);
  bf16* w1_b   = (bf16*)take((size_t)F_DIM * C_DIM * 2);
  bf16* w2_b   = (bf16*)take((size_t)C_DIM * F_DIM * 2);
  bf16* y1     = (bf16*)take((size_t)M_DIM * C_DIM * 2);
  bf16* qbuf   = (bf16*)take((size_t)M_DIM * C_DIM * 2);
  bf16* kbuf   = (bf16*)take((size_t)M_DIM * C_DIM * 2);
  bf16* vbuf   = (bf16*)take((size_t)M_DIM * C_DIM * 2);
  bf16* obuf   = (bf16*)take((size_t)M_DIM * C_DIM * 2);
  float* x2    = (float*)take((size_t)M_DIM * C_DIM * 4);
  bf16* y2     = (bf16*)take((size_t)M_DIM * C_DIM * 2);
  bf16* hbuf   = (bf16*)take((size_t)M_DIM * F_DIM * 2);

  // 1) weights -> bf16
  int nq = 3 * C_DIM * C_DIM, no = C_DIM * C_DIM, n1 = F_DIM * C_DIM, n2 = C_DIM * F_DIM;
  cvt_kernel<<<(nq + 255) / 256, 256, 0, stream>>>(w_qkv, wqkv_b, nq);
  cvt_kernel<<<(no + 255) / 256, 256, 0, stream>>>(w_o, wo_b, no);
  cvt_kernel<<<(n1 + 255) / 256, 256, 0, stream>>>(w1, w1_b, n1);
  cvt_kernel<<<(n2 + 255) / 256, 256, 0, stream>>>(w2, w2_b, n2);

  // 2) LN1
  ln_kernel<<<M_DIM / 8, 256, 0, stream>>>(x, ln1_g, ln1_b, y1);

  // 3) QKV projection + scatter (M=8192, N=768, K=256)
  gemm_bf16<0><<<dim3(768 / 64, M_DIM / 64), 128, 0, stream>>>(
      y1, wqkv_b, b_qkv, nullptr, nullptr, nullptr, qbuf, kbuf, vbuf, 768, C_DIM);

  // 4) flash attention with edge bias (4096 wave-tasks, 4 per block)
  attn_kernel<<<1024, 128, 0, stream>>>(qbuf, kbuf, vbuf, edge, kpm, obuf);

  // 5) output projection + residual -> x2 (M=8192, N=256, K=256)
  gemm_bf16<1><<<dim3(C_DIM / 64, M_DIM / 64), 128, 0, stream>>>(
      obuf, wo_b, b_o, x, x2, nullptr, nullptr, nullptr, nullptr, C_DIM, C_DIM);

  // 6) LN2
  ln_kernel<<<M_DIM / 8, 256, 0, stream>>>(x2, ln2_g, ln2_b, y2);

  // 7) FFN1 + exact GELU (M=8192, N=512, K=256)
  gemm_bf16<2><<<dim3(F_DIM / 64, M_DIM / 64), 128, 0, stream>>>(
      y2, w1_b, b1, nullptr, nullptr, hbuf, nullptr, nullptr, nullptr, F_DIM, C_DIM);

  // 8) FFN2 + residual -> out (M=8192, N=256, K=512)
  gemm_bf16<1><<<dim3(C_DIM / 64, M_DIM / 64), 128, 0, stream>>>(
      hbuf, w2_b, b2, x2, out, nullptr, nullptr, nullptr, nullptr, C_DIM, F_DIM);

  (void)in_sizes; (void)n_in; (void)out_size; (void)ws_size;
}
